// RecurrentSpaceNet_57775900066127
// MI455X (gfx1250) — matrix operand order repaired and checked by reading, not compiled, not run on hardware
//
#include <hip/hip_runtime.h>
#include <hip/hip_bf16.h>
#include <cstdint>

typedef __attribute__((ext_vector_type(16))) _Float16 v16h;
typedef __attribute__((ext_vector_type(8)))  _Float16 v8h;
typedef __attribute__((ext_vector_type(8)))  float    v8f;

union HFrag { v16h v; v8h h[2]; };

#define B_SZ 128
#define T_SZ 1024
#define NOUT 256

// ---------------------------------------------------------------------------
// Phase 1: ReLU RNN scan. 8 blocks x 512 threads (16 waves). Each block owns a
// 16-row batch slice; wave w owns output columns [16w, 16w+16). W_hh^T column
// slice lives in registers as 8 WMMA B-fragments for the whole time loop; h is
// a ping-pong f16 LDS buffer (one barrier per step). All 8 A-fragments are
// preloaded per step (single dscnt wait), and the wmma chain is split across
// two accumulators to shorten the dependency chain.
// ---------------------------------------------------------------------------
__global__ __launch_bounds__(512) void rnn_kernel(
    const float* __restrict__ x,  const float* __restrict__ W1,
    const float* __restrict__ b1, const float* __restrict__ W2,
    const float* __restrict__ b2, const float* __restrict__ W_ih,
    const float* __restrict__ W_hh, float* __restrict__ p_out)
{
  __shared__ alignas(16) _Float16 hb[2][16][264];   // 264 = 256 + 8 pad (bank-spread)
  __shared__ float xs[2][16][3];
  __shared__ float hid[64];

  const int tid    = threadIdx.x;
  const int lane   = tid & 31;
  const int wv     = tid >> 5;
  const int r0     = blockIdx.x * 16;               // batch-row base
  const int lrow   = lane & 15;
  const int hiHalf = (lane & 16) ? 1 : 0;
  const int n      = wv * 16 + lrow;                // this lane's output column

  // W_ih row for column n (3 inputs)
  const float wi0 = W_ih[n * 3 + 0];
  const float wi1 = W_ih[n * 3 + 1];
  const float wi2 = W_ih[n * 3 + 2];

  // Resident B fragments: B[k][n] = W_hh[n][k]. Lane holds 16 consecutive K of
  // row n (lanes 0-15: K 0..15 of chunk; lanes 16-31: K 16..31).
  HFrag bf[8];
#pragma unroll
  for (int kb = 0; kb < 8; ++kb) {
    const float* src = W_hh + (size_t)n * 256 + kb * 32 + (hiHalf ? 16 : 0);
#pragma unroll
    for (int i = 0; i < 16; ++i) bf[kb].v[i] = (_Float16)src[i];
  }

  // h0: identical across batch rows (input is ones).
  if (tid < 64) hid[tid] = fmaxf(W1[tid * 2] + W1[tid * 2 + 1] + b1[tid], 0.f);
  __syncthreads();
  if (tid < 256) {
    float s = b2[tid];
#pragma unroll 8
    for (int j = 0; j < 64; ++j) s += W2[tid * 64 + j] * hid[j];
    _Float16 hv = (_Float16)fmaxf(s, 0.f);
    for (int m = 0; m < 16; ++m) hb[0][m][tid] = hv;
  }
  if (tid < 48) {                                   // stage x for t = 0
    int m = tid / 3, c = tid % 3;
    xs[0][m][c] = x[(size_t)(r0 + m) * T_SZ * 3 + c];
  }
  __syncthreads();

  int cur = 0;
  const int koffA = hiHalf ? 16 : 0;

  for (int t = 0; t < T_SZ; ++t) {
    const int nxt = cur ^ 1;
    if (tid < 48 && t + 1 < T_SZ) {                 // stage next x early
      int m = tid / 3, c = tid % 3;
      xs[nxt][m][c] = x[(size_t)(r0 + m) * T_SZ * 3 + (size_t)(t + 1) * 3 + c];
    }

    // Preload all 8 A fragments (lane = row, K = {0..15} | {16..31})
    HFrag a[8];
#pragma unroll
    for (int kb = 0; kb < 8; ++kb) {
      a[kb].h[0] = *(const v8h*)&hb[cur][lrow][kb * 32 + koffA];
      a[kb].h[1] = *(const v8h*)&hb[cur][lrow][kb * 32 + 8 + koffA];
    }

    v8f acc0 = {};
    v8f acc1 = {};
#pragma unroll
    for (int kb = 0; kb < 8; kb += 2) {
      acc0 = __builtin_amdgcn_wmma_f32_16x16x32_f16(false, a[kb].v,     false, bf[kb].v,
                                                    (short)0, acc0, false, false);
      acc1 = __builtin_amdgcn_wmma_f32_16x16x32_f16(false, a[kb + 1].v, false, bf[kb + 1].v,
                                                    (short)0, acc1, false, false);
    }

#pragma unroll
    for (int i = 0; i < 8; ++i) {                   // C layout: vgpr i -> row
      int row = i + (hiHalf ? 8 : 0);
      float xpv = wi0 * xs[cur][row][0] + wi1 * xs[cur][row][1] + wi2 * xs[cur][row][2];
      float hv  = fmaxf(acc0[i] + acc1[i] + xpv, 0.f);
      p_out[((size_t)(r0 + row) * T_SZ + t) * NOUT + n] = hv;
      hb[nxt][row][n] = (_Float16)hv;
    }
    __syncthreads();
    cur = nxt;
  }
}

// ---------------------------------------------------------------------------
// Phase 1.5: sq[b*T + t] = sum_n p^2  (one wave per row, shuffle reduction)
// ---------------------------------------------------------------------------
__global__ __launch_bounds__(256) void sq_kernel(const float* __restrict__ p,
                                                 float* __restrict__ sq)
{
  const int lane = threadIdx.x & 31;
  const int row  = blockIdx.x * 8 + (threadIdx.x >> 5);
  const float* pr = p + (size_t)row * NOUT;
  float s = 0.f;
#pragma unroll
  for (int k = 0; k < 8; ++k) { float v = pr[lane + 32 * k]; s += v * v; }
#pragma unroll
  for (int off = 16; off > 0; off >>= 1) s += __shfl_xor(s, off, 32);
  if (lane == 0) sq[row] = s;
}

// ---------------------------------------------------------------------------
// Phase 2: corr = exp(-max(||pt||^2 + ||ps||^2 - 2 pt.ps, 0)).
// 512-thread block computes a 128x128 tile of corr for one batch; 16 waves,
// each owning a 2x2 group of 16x16 WMMA tiles. K staged in 32-wide f16 chunks
// through LDS (f32->f16 convert on load; p fits in L2 so re-reads are cheap).
// corr is write-once streamed output -> non-temporal stores keep p in L2.
// ---------------------------------------------------------------------------
__global__ __launch_bounds__(512) void gram_kernel(const float* __restrict__ p,
                                                   const float* __restrict__ sq,
                                                   float* __restrict__ corr)
{
  __shared__ alignas(16) _Float16 As[128 * 40];    // 40 = 32 + 8 pad
  __shared__ alignas(16) _Float16 Bs[128 * 40];
  __shared__ float sqa[128], sqb[128];

  const int tid    = threadIdx.x;
  const int lane   = tid & 31;
  const int wv     = tid >> 5;
  const int lrow   = lane & 15;
  const int hiHalf = (lane & 16) ? 1 : 0;
  const int b  = blockIdx.z;
  const int i0 = blockIdx.x * 128;
  const int j0 = blockIdx.y * 128;

  if (tid < 128)      sqa[tid]       = sq[(size_t)b * T_SZ + i0 + tid];
  else if (tid < 256) sqb[tid - 128] = sq[(size_t)b * T_SZ + j0 + (tid - 128)];

  const int m0 = (wv >> 2) * 2;                     // tile rows (of 8)
  const int n0 = (wv & 3) * 2;                      // tile cols (of 8)

  v8f zero = {};
  v8f acc[2][2];
  acc[0][0] = zero; acc[0][1] = zero; acc[1][0] = zero; acc[1][1] = zero;

  const int sr = (tid * 8) >> 5;                    // staging: 8 floats/thread
  const int sc = (tid * 8) & 31;
  const float* srcA = p + ((size_t)b * T_SZ + i0 + sr) * NOUT + sc;
  const float* srcB = p + ((size_t)b * T_SZ + j0 + sr) * NOUT + sc;

  for (int kb = 0; kb < 8; ++kb) {
    __syncthreads();                                // prev chunk fully consumed
    {
      const float* a4 = srcA + kb * 32;
      const float* b4 = srcB + kb * 32;
      float4 fa0 = *(const float4*)a4;  float4 fa1 = *(const float4*)(a4 + 4);
      float4 fb0 = *(const float4*)b4;  float4 fb1 = *(const float4*)(b4 + 4);
      v8h ha, hbv;
      ha[0] = (_Float16)fa0.x; ha[1] = (_Float16)fa0.y; ha[2] = (_Float16)fa0.z; ha[3] = (_Float16)fa0.w;
      ha[4] = (_Float16)fa1.x; ha[5] = (_Float16)fa1.y; ha[6] = (_Float16)fa1.z; ha[7] = (_Float16)fa1.w;
      hbv[0] = (_Float16)fb0.x; hbv[1] = (_Float16)fb0.y; hbv[2] = (_Float16)fb0.z; hbv[3] = (_Float16)fb0.w;
      hbv[4] = (_Float16)fb1.x; hbv[5] = (_Float16)fb1.y; hbv[6] = (_Float16)fb1.z; hbv[7] = (_Float16)fb1.w;
      *(v8h*)&As[sr * 40 + sc] = ha;
      *(v8h*)&Bs[sr * 40 + sc] = hbv;
    }
    __syncthreads();

    HFrag a[2], bb[2];
#pragma unroll
    for (int mi = 0; mi < 2; ++mi) {
      const _Float16* ap = &As[(16 * (m0 + mi) + lrow) * 40];
      a[mi].h[0] = *(const v8h*)(ap + (hiHalf ? 16 : 0));
      a[mi].h[1] = *(const v8h*)(ap + 8 + (hiHalf ? 16 : 0));
    }
#pragma unroll
    for (int ni = 0; ni < 2; ++ni) {                // B = p^T: lane = column
      const _Float16* bp = &Bs[(16 * (n0 + ni) + lrow) * 40 + (hiHalf ? 16 : 0)];
      bb[ni].h[0] = *(const v8h*)bp;
      bb[ni].h[1] = *(const v8h*)(bp + 8);
    }
#pragma unroll
    for (int mi = 0; mi < 2; ++mi)
#pragma unroll
      for (int ni = 0; ni < 2; ++ni)
        acc[mi][ni] = __builtin_amdgcn_wmma_f32_16x16x32_f16(
            false, a[mi].v, false, bb[ni].v, (short)0, acc[mi][ni], false, false);
  }

#pragma unroll
  for (int mi = 0; mi < 2; ++mi) {
#pragma unroll
    for (int ni = 0; ni < 2; ++ni) {
#pragma unroll
      for (int i = 0; i < 8; ++i) {
        int row = 16 * (m0 + mi) + i + (hiHalf ? 8 : 0);
        int col = 16 * (n0 + ni) + lrow;
        float dp = sqa[row] + sqb[col] - 2.f * acc[mi][ni][i];
        dp = fmaxf(dp, 0.f);
        // streamed write-once output: non-temporal so p/sq stay L2-resident
        __builtin_nontemporal_store(__expf(-dp),
            &corr[((size_t)b * T_SZ + i0 + row) * T_SZ + j0 + col]);
      }
    }
  }
}

// ---------------------------------------------------------------------------
extern "C" void kernel_launch(void* const* d_in, const int* in_sizes, int n_in,
                              void* d_out, int out_size, void* d_ws, size_t ws_size,
                              hipStream_t stream)
{
  (void)in_sizes; (void)n_in; (void)out_size; (void)ws_size;
  const float* x    = (const float*)d_in[0];
  const float* W1   = (const float*)d_in[1];
  const float* b1   = (const float*)d_in[2];
  const float* W2   = (const float*)d_in[3];
  const float* b2   = (const float*)d_in[4];
  const float* W_ih = (const float*)d_in[5];
  const float* W_hh = (const float*)d_in[6];

  float* corr  = (float*)d_out;                          // [B,T,T]
  float* p_out = corr + (size_t)B_SZ * T_SZ * T_SZ;      // [B,T,256]
  float* sq    = (float*)d_ws;                           // [B*T] floats

  rnn_kernel<<<dim3(B_SZ / 16), 512, 0, stream>>>(x, W1, b1, W2, b2, W_ih, W_hh, p_out);
  sq_kernel<<<dim3((B_SZ * T_SZ) / 8), 256, 0, stream>>>(p_out, sq);
  gram_kernel<<<dim3(T_SZ / 128, T_SZ / 128, B_SZ), 512, 0, stream>>>(p_out, sq, corr);
}
